// MultiheadAttention_75754633167151
// MI455X (gfx1250) — compile-verified, matrix operand order checked by reading, not run on hardware
//
#include <hip/hip_runtime.h>
#include <hip/hip_bf16.h>

typedef _Float16 f16;
typedef __attribute__((ext_vector_type(16))) _Float16 v16h;
typedef __attribute__((ext_vector_type(8)))  _Float16 v8h;
typedef __attribute__((ext_vector_type(4)))  _Float16 v4h;
typedef __attribute__((ext_vector_type(8)))  float    v8f;
typedef __attribute__((ext_vector_type(4)))  float    v4f;
typedef __attribute__((ext_vector_type(4)))  unsigned int u32x4;
typedef __attribute__((ext_vector_type(8)))  unsigned int u32x8;

constexpr int kB = 4, kT = 2048, kC = 2048, kH = 16, kD = 128;
constexpr int kM = kB * kT;  // 8192 rows in all GEMMs

__device__ __forceinline__ v16h mk16(v8h lo, v8h hi) {
  v16h r;
#pragma unroll
  for (int i = 0; i < 8; ++i) { r[i] = lo[i]; r[i + 8] = hi[i]; }
  return r;
}

// Async 16B global -> LDS copy (ASYNCcnt-tracked, no VGPR round trip).
__device__ __forceinline__ void async_copy_b128(const void* gsrc, void* ldst) {
  asm volatile("global_load_async_to_lds_b128 %0, %1, off"
               :: "v"((unsigned int)(unsigned long long)ldst),
                  "v"((unsigned long long)gsrc)
               : "memory");
}
__device__ __forceinline__ void wait_async0() {
  asm volatile("s_wait_asynccnt 0" ::: "memory");
}

// Tensor Data Mover: 2D tile (rows x cols f16 elements) from global (row
// stride in halves) into LDS, padded by padHalves per row (pad must be a
// multiple of 8 halves; cols*2 must be a multiple-of-16 bytes).
// Issued by one wave; track with TENSORcnt.
__device__ __forceinline__ void tdm_load_2d_f16(const f16* gsrc, void* ldst,
                                                unsigned rows, unsigned cols,
                                                unsigned rowStrideHalves,
                                                unsigned padHalves) {
  unsigned long long ga = (unsigned long long)gsrc;
  u32x4 g0;
  g0[0] = 1u;  // count=1, no gather, user descriptor
  g0[1] = (unsigned int)(unsigned long long)ldst;               // lds_addr
  g0[2] = (unsigned int)ga;                                     // global[31:0]
  g0[3] = (unsigned int)((ga >> 32) & 0x1FFFFFFu) | (2u << 30); // global[56:32] | type=2
  // pad_interval code: interval of (2<<code) DWORDs == one row of cols halves
  unsigned rowDW = cols >> 1;           // cols*2B / 4
  unsigned icode = 0;
  for (unsigned v = 2; v < rowDW; v <<= 1) ++icode;
  unsigned padDW = padHalves >> 1;
  u32x8 g1;
  g1[0] = (1u << 16)                    // data_size = 2 bytes
        | ((padHalves ? 1u : 0u) << 20) // pad_enable
        | (icode << 22)                 // pad_interval
        | ((padDW ? padDW - 1u : 0u) << 25);  // pad_amount (code = DWORDs-1)
  g1[1] = (cols & 0xFFFFu) << 16;       // tensor_dim0[15:0] -> bits 63:48
  g1[2] = (cols >> 16) | ((rows & 0xFFFFu) << 16);  // dim0 hi | tensor_dim1 lo
  g1[3] = (rows >> 16) | ((cols & 0xFFFFu) << 16);  // dim1 hi | tile_dim0
  g1[4] = rows & 0xFFFFu;               // tile_dim1 (tile_dim2 = 0)
  g1[5] = rowStrideHalves;              // tensor_dim0_stride[31:0]
  g1[6] = 0u;                           // stride hi | tensor_dim1_stride lo
  g1[7] = 0u;
  asm volatile("tensor_load_to_lds %0, %1" :: "s"(g0), "s"(g1) : "memory");
}

// ---------------------------------------------------------------------------
// Elementwise f32 -> f16 conversion (8 elements/thread, b128 in / b128 out).
// ---------------------------------------------------------------------------
__global__ __launch_bounds__(256) void cvt_f32_f16_kernel(
    const float* __restrict__ src, f16* __restrict__ dst) {
  size_t i = ((size_t)blockIdx.x * 256 + threadIdx.x) * 8;
  v4f a = *(const v4f*)(src + i);
  v4f b = *(const v4f*)(src + i + 4);
  v8h o;
  o[0] = (f16)a[0]; o[1] = (f16)a[1]; o[2] = (f16)a[2]; o[3] = (f16)a[3];
  o[4] = (f16)b[0]; o[5] = (f16)b[1]; o[6] = (f16)b[2]; o[7] = (f16)b[3];
  *(v8h*)(dst + i) = o;
}

// ---------------------------------------------------------------------------
// Tiled GEMM: Out[M,N] = A[M,K] @ W[K,N] + bias[N]
// Block tile 128x256, K-step 32, 256 threads = 8 waves (2x4); each wave
// computes a 64x64 sub-tile as 4x4 WMMA_F32_16X16X32_F16 accumulators
// (16 WMMAs per 8 fragment loads per K-step).
// A f16 path: tile staged with global_load_async_to_lds_b128.
// A f32 path: sync load + convert. W always f32, staged TRANSPOSED in LDS.
// ---------------------------------------------------------------------------
template <bool A_F16, bool OUT_F16>
__global__ __launch_bounds__(256) void gemm_bias_kernel(
    const void* __restrict__ Ap, const float* __restrict__ W,
    const float* __restrict__ bias, void* __restrict__ Outp,
    int M, int N, int K) {
  constexpr int BM = 128, BN = 256, BK = 32, LDS_S = 48;  // 96B row stride (16B mult)
  __shared__ f16 As[BM * LDS_S];
  __shared__ f16 Bt[BN * LDS_S];  // Bt[n][k]

  const int tid  = threadIdx.x;
  const int lane = tid & 31;
  const int wave = tid >> 5;
  const int wm   = wave >> 2;   // 0..1  -> 64-row group
  const int wn   = wave & 3;    // 0..3  -> 64-col group
  const int l16  = lane & 15;
  const int ksh  = (lane < 16) ? 0 : 8;  // K split for upper lanes (ISA 16-bit layout)
  const int m0 = blockIdx.y * BM, n0 = blockIdx.x * BN;

  v8f acc[4][4];
#pragma unroll
  for (int mt = 0; mt < 4; ++mt)
#pragma unroll
    for (int nt = 0; nt < 4; ++nt) {
      v8f z = {0.f, 0.f, 0.f, 0.f, 0.f, 0.f, 0.f, 0.f};
      acc[mt][nt] = z;
    }

  for (int kb = 0; kb < K; kb += BK) {
    // ---- stage A tile (128x32 f16) ----
    if constexpr (A_F16) {
      const f16* A = (const f16*)Ap;
#pragma unroll
      for (int i = 0; i < 2; ++i) {
        int idx = tid + i * 256;   // 0..511 octets of halves
        int row = idx >> 2;        // 0..127
        int c8  = (idx & 3) * 8;   // 0..24
        async_copy_b128(A + (size_t)(m0 + row) * K + kb + c8,
                        As + row * LDS_S + c8);
      }
    } else {
      const float* A = (const float*)Ap;
#pragma unroll
      for (int i = 0; i < 4; ++i) {
        int idx = tid + i * 256;   // 0..1023 quads
        int row = idx >> 3;        // 0..127
        int c4  = (idx & 7) * 4;   // 0..28
        v4f fv = *(const v4f*)(A + (size_t)(m0 + row) * K + kb + c4);
        v4h hv;
        hv[0] = (f16)fv[0]; hv[1] = (f16)fv[1]; hv[2] = (f16)fv[2]; hv[3] = (f16)fv[3];
        *(v4h*)(As + row * LDS_S + c4) = hv;
      }
    }
    // ---- stage B tile (32x256) transposed into Bt[n][k], f32->f16 ----
#pragma unroll
    for (int i = 0; i < 8; ++i) {
      int idx = tid + i * 256;       // 0..2047 quads
      int kr  = idx >> 6;            // 0..31
      int c4  = (idx & 63) * 4;      // 0..252
      v4f fv = *(const v4f*)(W + (size_t)(kb + kr) * N + n0 + c4);
      Bt[(c4 + 0) * LDS_S + kr] = (f16)fv[0];
      Bt[(c4 + 1) * LDS_S + kr] = (f16)fv[1];
      Bt[(c4 + 2) * LDS_S + kr] = (f16)fv[2];
      Bt[(c4 + 3) * LDS_S + kr] = (f16)fv[3];
    }
    if constexpr (A_F16) wait_async0();
    __syncthreads();

    // ---- compute: fragments per documented 16-bit A/B layouts ----
    v16h bf[4];
#pragma unroll
    for (int nt = 0; nt < 4; ++nt) {
      const f16* p = Bt + (wn * 64 + nt * 16 + l16) * LDS_S + ksh;
      bf[nt] = mk16(*(const v8h*)p, *(const v8h*)(p + 16));
    }
#pragma unroll
    for (int mt = 0; mt < 4; ++mt) {
      const f16* p = As + (wm * 64 + mt * 16 + l16) * LDS_S + ksh;
      v16h af = mk16(*(const v8h*)p, *(const v8h*)(p + 16));
#pragma unroll
      for (int nt = 0; nt < 4; ++nt)
        acc[mt][nt] = __builtin_amdgcn_wmma_f32_16x16x32_f16(
            false, af, false, bf[nt], (short)0, acc[mt][nt], false, false);
    }
    __syncthreads();
  }

  // ---- epilogue: bias add + store (C layout: lanes 0-15 rows r, 16-31 rows r+8) ----
#pragma unroll
  for (int mt = 0; mt < 4; ++mt)
#pragma unroll
    for (int nt = 0; nt < 4; ++nt) {
      int gcol = n0 + wn * 64 + nt * 16 + l16;
      float bv = bias[gcol];
#pragma unroll
      for (int r = 0; r < 8; ++r) {
        int grow = m0 + wm * 64 + mt * 16 + ((lane < 16) ? r : r + 8);
        float o = acc[mt][nt][r] + bv;
        if constexpr (OUT_F16)
          ((f16*)Outp)[(size_t)grow * N + gcol] = (f16)o;
        else
          ((float*)Outp)[(size_t)grow * N + gcol] = o;
      }
    }
}

// ---------------------------------------------------------------------------
// Flash attention: grid (T/128, B*H), 256 threads = 8 waves.
// Wave w owns q rows [q0+16w, q0+16w+16). Q fragments in registers (16x128).
// K tiles staged by the Tensor Data Mover (one tensor_load_to_lds per KV
// step, TENSORcnt-tracked). V tiles staged sync + transposed.
// Online softmax, f32 accumulation.
// ---------------------------------------------------------------------------
__global__ __launch_bounds__(256) void flash_attn_kernel(
    const f16* __restrict__ Qh, const f16* __restrict__ Kh,
    const f16* __restrict__ Vh, f16* __restrict__ Oa,
    const int* __restrict__ is_causal_p) {
  constexpr int KVT = 64;
  constexpr int LKT = 136;  // 64x(128+8) halves, row stride 272B (16B mult)
  constexpr int LVT = 72;   // 128x(64+8) halves, row stride 144B (16B mult)
  constexpr int LPB = 72;   // per-wave P scratch 16x(64+8)
  __shared__ f16 Kt[64 * LKT];
  __shared__ f16 Vt[128 * LVT];
  __shared__ f16 Pb[8 * 16 * LPB];

  const int tid  = threadIdx.x;
  const int lane = tid & 31;
  const int wave = tid >> 5;
  const int l16  = lane & 15;
  const int ksh  = (lane < 16) ? 0 : 8;
  const int bh = blockIdx.y;
  const int b = bh / kH, h = bh % kH;
  const int q0 = blockIdx.x * 128;
  const int qr = q0 + wave * 16;
  const bool causal = (*is_causal_p) != 0;
  const float scale = 0.08838834764831845f;  // 1/sqrt(128)

  // Q fragments: 4 chunks of 16x32 f16 (A layout), loaded straight from global
  v16h qf[4];
  {
    const f16* qp = Qh + ((size_t)(b * kT + qr + l16)) * kC + h * kD + ksh;
#pragma unroll
    for (int c = 0; c < 4; ++c)
      qf[c] = mk16(*(const v8h*)(qp + c * 32), *(const v8h*)(qp + c * 32 + 16));
  }

  v8f Oacc[8];
  float mrow[8], lrow[8];
#pragma unroll
  for (int j = 0; j < 8; ++j) {
    v8f z = {0.f, 0.f, 0.f, 0.f, 0.f, 0.f, 0.f, 0.f};
    Oacc[j] = z;
    mrow[j] = -1e30f;
    lrow[j] = 0.f;
  }

  const int kvend = causal ? (q0 + 128) : kT;
  for (int kv0 = 0; kv0 < kvend; kv0 += KVT) {
    // ---- stage K tile (64x128) via Tensor Data Mover (wave 0 only) ----
    if (wave == 0) {
      tdm_load_2d_f16(Kh + ((size_t)(b * kT + kv0)) * kC + h * kD, Kt,
                      /*rows=*/KVT, /*cols=*/kD, /*rowStrideHalves=*/kC,
                      /*padHalves=*/LKT - kD);
    }
    // ---- stage V tile transposed: Vt[d][kv] ----
#pragma unroll
    for (int i = 0; i < 4; ++i) {
      int idx = tid + i * 256;
      int r  = idx >> 4;
      int c8 = (idx & 15) * 8;
      v8h v = *(const v8h*)(Vh + ((size_t)(b * kT + kv0 + r)) * kC + h * kD + c8);
#pragma unroll
      for (int j = 0; j < 8; ++j) Vt[(c8 + j) * LVT + r] = v[j];
    }
    if (wave == 0) __builtin_amdgcn_s_wait_tensorcnt(0);
    __syncthreads();

    // ---- S = Q K^T : 4 N-subtiles x 4 K-chunks = 16 WMMAs ----
    v8f s[4];
#pragma unroll
    for (int j = 0; j < 4; ++j) {
      v8f sj = {0.f, 0.f, 0.f, 0.f, 0.f, 0.f, 0.f, 0.f};
#pragma unroll
      for (int c = 0; c < 4; ++c) {
        const f16* kp = Kt + (j * 16 + l16) * LKT + c * 32 + ksh;
        v16h bf = mk16(*(const v8h*)kp, *(const v8h*)(kp + 16));
        sj = __builtin_amdgcn_wmma_f32_16x16x32_f16(false, qf[c], false, bf,
                                                    (short)0, sj, false, false);
      }
      s[j] = sj;
    }
    // ---- scale + causal mask ----
#pragma unroll
    for (int j = 0; j < 4; ++j)
#pragma unroll
      for (int r = 0; r < 8; ++r) {
        int qrow = qr + ((lane < 16) ? r : r + 8);
        int kcol = kv0 + j * 16 + l16;
        float v = s[j][r] * scale;
        if (causal && kcol > qrow) v = -1e30f;
        s[j][r] = v;
      }
    // ---- online softmax (row reductions across 16-lane halves) ----
    float mloc[8], rs[8], alpha[8];
#pragma unroll
    for (int r = 0; r < 8; ++r) {
      float mv = s[0][r];
#pragma unroll
      for (int j = 1; j < 4; ++j) mv = fmaxf(mv, s[j][r]);
      mloc[r] = mv;
    }
#pragma unroll
    for (int msk = 1; msk < 16; msk <<= 1)
#pragma unroll
      for (int r = 0; r < 8; ++r)
        mloc[r] = fmaxf(mloc[r], __shfl_xor(mloc[r], msk, 32));
#pragma unroll
    for (int r = 0; r < 8; ++r) {
      float mn = fmaxf(mrow[r], mloc[r]);
      alpha[r] = __expf(mrow[r] - mn);
      mrow[r] = mn;
      rs[r] = 0.f;
    }
#pragma unroll
    for (int j = 0; j < 4; ++j)
#pragma unroll
      for (int r = 0; r < 8; ++r) {
        float p = __expf(s[j][r] - mrow[r]);
        s[j][r] = p;
        rs[r] += p;
      }
#pragma unroll
    for (int msk = 1; msk < 16; msk <<= 1)
#pragma unroll
      for (int r = 0; r < 8; ++r) rs[r] += __shfl_xor(rs[r], msk, 32);
#pragma unroll
    for (int r = 0; r < 8; ++r) lrow[r] = lrow[r] * alpha[r] + rs[r];
#pragma unroll
    for (int jd = 0; jd < 8; ++jd)
#pragma unroll
      for (int r = 0; r < 8; ++r) Oacc[jd][r] *= alpha[r];

    // ---- P: C layout -> A layout via per-wave LDS scratch ----
#pragma unroll
    for (int r = 0; r < 8; ++r) {
      int rowM = (lane < 16) ? r : r + 8;
#pragma unroll
      for (int j = 0; j < 4; ++j)
        Pb[(wave * 16 + rowM) * LPB + j * 16 + l16] = (f16)s[j][r];
    }
    asm volatile("s_wait_dscnt 0" ::: "memory");  // wave-local LDS RAW fence
    v16h pf[2];
#pragma unroll
    for (int c = 0; c < 2; ++c) {
      const f16* pp = Pb + (wave * 16 + l16) * LPB + c * 32 + ksh;
      pf[c] = mk16(*(const v8h*)pp, *(const v8h*)(pp + 16));
    }
    // ---- O += P V : 8 D-subtiles x 2 KV-chunks = 16 WMMAs ----
#pragma unroll
    for (int jd = 0; jd < 8; ++jd)
#pragma unroll
      for (int c = 0; c < 2; ++c) {
        const f16* vp = Vt + (jd * 16 + l16) * LVT + c * 32 + ksh;
        v16h bf = mk16(*(const v8h*)vp, *(const v8h*)(vp + 16));
        Oacc[jd] = __builtin_amdgcn_wmma_f32_16x16x32_f16(
            false, pf[c], false, bf, (short)0, Oacc[jd], false, false);
      }
    __syncthreads();
  }

  // ---- epilogue: normalize by l, store f16 attn output ----
#pragma unroll
  for (int jd = 0; jd < 8; ++jd)
#pragma unroll
    for (int r = 0; r < 8; ++r) {
      int rowM = (lane < 16) ? r : r + 8;
      size_t off = ((size_t)(b * kT + qr + rowM)) * kC + h * kD + jd * 16 + l16;
      Oa[off] = (f16)(Oacc[jd][r] / lrow[r]);
    }
}

// ---------------------------------------------------------------------------
// Launcher.
// Inputs: 0 querys,1 keys,2 values,3 Wq,4 bq,5 Wk,6 bk,7 Wv,8 bv,9 Wo,10 bo,11 is_causal
// ws layout: Q,K,V,attn f16 (128 MB) [+ optional xq,xk,xv f16 copies, 96 MB]
// ---------------------------------------------------------------------------
extern "C" void kernel_launch(void* const* d_in, const int* in_sizes, int n_in,
                              void* d_out, int out_size, void* d_ws, size_t ws_size,
                              hipStream_t stream) {
  (void)in_sizes; (void)n_in; (void)out_size;
  const size_t elems = (size_t)kM * kC;
  f16* qws = (f16*)d_ws;
  f16* kws = qws + elems;
  f16* vws = kws + elems;
  f16* aws = vws + elems;
  f16* xq  = aws + elems;
  f16* xk  = xq + elems;
  f16* xv  = xk + elems;
  const bool pre = ws_size >= 7 * elems * sizeof(f16);

  dim3 gg(kC / 256, kM / 128);  // (8, 64)
  if (pre) {
    // One-time f32->f16 conversion of activations: halves A-operand re-read
    // traffic in the projection GEMMs and enables async LDS staging.
    const int cvt_blocks = (int)(elems / (256 * 8));
    cvt_f32_f16_kernel<<<cvt_blocks, 256, 0, stream>>>((const float*)d_in[0], xq);
    cvt_f32_f16_kernel<<<cvt_blocks, 256, 0, stream>>>((const float*)d_in[1], xk);
    cvt_f32_f16_kernel<<<cvt_blocks, 256, 0, stream>>>((const float*)d_in[2], xv);
    gemm_bias_kernel<true, true><<<gg, 256, 0, stream>>>(
        xq, (const float*)d_in[3], (const float*)d_in[4], qws, kM, kC, kC);
    gemm_bias_kernel<true, true><<<gg, 256, 0, stream>>>(
        xk, (const float*)d_in[5], (const float*)d_in[6], kws, kM, kC, kC);
    gemm_bias_kernel<true, true><<<gg, 256, 0, stream>>>(
        xv, (const float*)d_in[7], (const float*)d_in[8], vws, kM, kC, kC);
  } else {
    gemm_bias_kernel<false, true><<<gg, 256, 0, stream>>>(
        d_in[0], (const float*)d_in[3], (const float*)d_in[4], qws, kM, kC, kC);
    gemm_bias_kernel<false, true><<<gg, 256, 0, stream>>>(
        d_in[1], (const float*)d_in[5], (const float*)d_in[6], kws, kM, kC, kC);
    gemm_bias_kernel<false, true><<<gg, 256, 0, stream>>>(
        d_in[2], (const float*)d_in[7], (const float*)d_in[8], vws, kM, kC, kC);
  }

  dim3 fg(kT / 128, kB * kH);  // (16, 64)
  flash_attn_kernel<<<fg, 256, 0, stream>>>(qws, kws, vws, aws,
                                            (const int*)d_in[11]);

  gemm_bias_kernel<true, false><<<gg, 256, 0, stream>>>(
      aws, (const float*)d_in[9], (const float*)d_in[10], d_out, kM, kC, kC);
}
